// LGANLayer_14851996909629
// MI455X (gfx1250) — compile-verified, matrix operand order checked by reading, not run on hardware
//
#include <hip/hip_runtime.h>
#include <hip/hip_bf16.h>

// ---------------------------------------------------------------------------
// LGAN layer for MI455X (gfx1250, wave32, WMMA).
//  - Dense MLPs: V_WMMA_F32_16X16X32_F16, f32 accum.
//  - Weights pre-swizzled to f16 fragment order: B fragment = 2x global b128.
//  - Activations staged in LDS as packed f16: A fragment = 2x ds b128.
//  - K-loop software-pipelined: fragments for kt+1 are loaded before the
//    WMMAs of kt issue, keeping ~10 b128 loads in flight at all times.
//  - Segment sums: f32 global atomics (scatter targets are L2-resident).
// ---------------------------------------------------------------------------

#define D_   128
#define MH_  256

typedef __attribute__((ext_vector_type(16))) _Float16     v16h;
typedef __attribute__((ext_vector_type(8)))  float        v8f;
typedef __attribute__((ext_vector_type(4)))  unsigned int v4u;

// K offset inside a 32-wide K tile for dword index i (0..7) and half-wave.
// Per CDNA5 ISA 16-bit A-matrix 16x32 layout (CDNA5 ISA §7.12.2).
__device__ __forceinline__ int wmma_kmap(int i, int half) {
    return (i < 4) ? (half * 8 + 2 * i) : (16 + half * 8 + 2 * (i - 4));
}

__device__ __forceinline__ unsigned int pkh(float lo, float hi) {
    union { _Float16 h[2]; unsigned int u; } un;
    un.h[0] = (_Float16)lo;
    un.h[1] = (_Float16)hi;
    return un.u;
}

union frag_cast { struct { v4u lo, hi; } u; v16h h; };

// A fragment (16x32 f16) from packed-f16 LDS tile (row stride ldDW dwords,
// multiple of 4). Two ds_load_b128 per lane, bank-conflict free (ldDW%64==4m).
__device__ __forceinline__ v16h load_a_lds16(const unsigned int* X, int ldDW,
                                             int kt, int lane) {
    const int m = lane & 15, half = lane >> 4;
    const unsigned int* p = X + m * ldDW + kt * 16 + half * 4;
    frag_cast f;
    f.u.lo = *(const v4u*)p;
    f.u.hi = *(const v4u*)(p + 8);
    return f.h;
}

// B fragment (32x16 f16) from pre-swizzled weight buffer:
// dwords laid out as [kt][nt][lane][i]; 8 contiguous dwords per lane.
__device__ __forceinline__ v16h load_b_sw(const unsigned int* __restrict__ Wsw,
                                          int ntTiles, int kt, int nt, int lane) {
    const unsigned int* p = Wsw + ((((size_t)kt * ntTiles + nt) * 32) + lane) * 8;
    frag_cast f;
    f.u.lo = *(const v4u*)p;
    f.u.hi = *(const v4u*)(p + 4);
    return f.h;
}

__device__ __forceinline__ v8f wmma_acc(v16h a, v16h b, v8f c) {
    return __builtin_amdgcn_wmma_f32_16x16x32_f16(false, a, false, b,
                                                  (short)0, c, false, false);
}

// Software-pipelined K-loop of a tiled GEMM: fragments for step kt+1 are
// loaded into a second register set before the WMMAs of step kt.
template <int NT, int KT>
__device__ __forceinline__ void gemm_loop(v8f (&acc)[NT],
                                          const unsigned int* Xlds, int ldDW,
                                          const unsigned int* __restrict__ Wsw,
                                          int ntTiles, int ntBase, int lane) {
    v16h a = load_a_lds16(Xlds, ldDW, 0, lane);
    v16h b[NT];
#pragma unroll
    for (int t = 0; t < NT; ++t) b[t] = load_b_sw(Wsw, ntTiles, 0, ntBase + t, lane);
#pragma unroll
    for (int kt = 0; kt < KT; ++kt) {
        v16h an;
        v16h bn[NT];
        if (kt + 1 < KT) {
            an = load_a_lds16(Xlds, ldDW, kt + 1, lane);
#pragma unroll
            for (int t = 0; t < NT; ++t)
                bn[t] = load_b_sw(Wsw, ntTiles, kt + 1, ntBase + t, lane);
        } else {
            an = a;
#pragma unroll
            for (int t = 0; t < NT; ++t) bn[t] = b[t];
        }
#pragma unroll
        for (int t = 0; t < NT; ++t) acc[t] = wmma_acc(a, b[t], acc[t]);
        a = an;
#pragma unroll
        for (int t = 0; t < NT; ++t) b[t] = bn[t];
    }
}

// ---------------------------------------------------------------------------
// Prep: swizzle f32 weight W[K][N] into f16 WMMA-fragment order.
// Output dword index = ((kt*NT + nt)*32 + lane)*8 + i.
// ---------------------------------------------------------------------------
__global__ void swizzle_weight_kernel(const float* __restrict__ W, int K, int N,
                                      unsigned int* __restrict__ Wsw) {
    const int idx = blockIdx.x * blockDim.x + threadIdx.x;
    const int NT = N >> 4;
    const int total = (K >> 5) * NT * 32 * 8;
    if (idx >= total) return;
    const int i    = idx & 7;
    const int lane = (idx >> 3) & 31;
    const int fid  = idx >> 8;
    const int kt = fid / NT, nt = fid % NT;
    const int half = lane >> 4;
    const int k = kt * 32 + wmma_kmap(i, half);
    const int n = nt * 16 + (lane & 15);
    Wsw[idx] = pkh(W[(size_t)k * N + n], W[(size_t)(k + 1) * N + n]);
}

// ---------------------------------------------------------------------------
// Kernel 1: per-edge MLP   h_e = relu((h[u]+h[v]) @ We1 + be1) @ We2 + be2
// One workgroup (128 thr = 4 waves) per 16-edge tile.
// ---------------------------------------------------------------------------
__global__ void __launch_bounds__(128)
edge_mlp_kernel(const float* __restrict__ h, const int* __restrict__ edges, int E,
                const unsigned int* __restrict__ W1s, const float* __restrict__ b1,
                const unsigned int* __restrict__ W2s, const float* __restrict__ b2,
                float* __restrict__ he) {
    const int LDX = D_ / 2 + 4;    // 68 dwords  (16 x 128 f16 tile)
    const int LDH = MH_ / 2 + 4;   // 132 dwords (16 x 256 f16 tile)
    __shared__ __align__(16) unsigned int Xs[16 * (D_ / 2 + 4)];
    __shared__ __align__(16) unsigned int Hs[16 * (MH_ / 2 + 4)];

    const int e0   = blockIdx.x * 16;
    const int tid  = threadIdx.x;
    const int lane = tid & 31;
    const int wave = tid >> 5;

    // Stage X = f16(h[u] + h[v]), packed 2 cols/dword.
    for (int idx = tid; idx < 16 * (D_ / 2); idx += 128) {
        const int r = idx >> 6, cd = idx & (D_ / 2 - 1);
        const int e = e0 + r;
        unsigned int pv = 0u;
        if (e < E) {
            const int u = edges[2 * e], v = edges[2 * e + 1];
            const float2 a = ((const float2*)h)[(size_t)u * (D_ / 2) + cd];
            const float2 b = ((const float2*)h)[(size_t)v * (D_ / 2) + cd];
            pv = pkh(a.x + b.x, a.y + b.y);
        }
        Xs[r * LDX + cd] = pv;
    }
    __syncthreads();

    const int n = lane & 15, half = lane >> 4;

    // Layer 1: hidden = relu(X @ W1 + b1)  (16x256, K=128)
    {
        v8f acc[4] = {};
        gemm_loop<4, 4>(acc, Xs, LDX, W1s, MH_ / 16, wave * 4, lane);
        _Float16* Hh = (_Float16*)Hs;
#pragma unroll
        for (int t = 0; t < 4; ++t) {
            const int n0 = (wave * 4 + t) * 16;
            const float bias = b1[n0 + n];
#pragma unroll
            for (int r = 0; r < 8; ++r) {
                const int m = r + 8 * half;
                const float v = acc[t][r] + bias;
                Hh[m * (LDH * 2) + n0 + n] = (_Float16)(v > 0.f ? v : 0.f);
            }
        }
    }
    __syncthreads();

    // Layer 2: h_e = hidden @ W2 + b2  (16x128, K=256)
    {
        v8f acc[2] = {};
        gemm_loop<2, 8>(acc, Hs, LDH, W2s, D_ / 16, wave * 2, lane);
#pragma unroll
        for (int t = 0; t < 2; ++t) {
            const int n0 = (wave * 2 + t) * 16;
            const float bias = b2[n0 + n];
#pragma unroll
            for (int r = 0; r < 8; ++r) {
                const int m = r + 8 * half;
                const int e = e0 + m;
                if (e < E) he[(size_t)e * D_ + n0 + n] = acc[t][r] + bias;
            }
        }
    }
}

// ---------------------------------------------------------------------------
// Kernel 2: aggr_t[u] += h_e, aggr_t[v] += h_e, deg[u]++, deg[v]++
// ---------------------------------------------------------------------------
__global__ void aggr_edges_kernel(const float4* __restrict__ he4,
                                  const int* __restrict__ edges, int E,
                                  float* __restrict__ aggr_t, float* __restrict__ deg) {
    const long idx = (long)blockIdx.x * blockDim.x + threadIdx.x;
    if (idx >= (long)E * (D_ / 4)) return;
    const int e  = (int)(idx >> 5);
    const int c4 = (int)(idx & 31);
    const float4 v = he4[idx];
    const int u = edges[2 * e], w = edges[2 * e + 1];
    float* pu = &aggr_t[(size_t)u * D_ + c4 * 4];
    float* pw = &aggr_t[(size_t)w * D_ + c4 * 4];
    atomicAdd(pu + 0, v.x); atomicAdd(pu + 1, v.y);
    atomicAdd(pu + 2, v.z); atomicAdd(pu + 3, v.w);
    atomicAdd(pw + 0, v.x); atomicAdd(pw + 1, v.y);
    atomicAdd(pw + 2, v.z); atomicAdd(pw + 3, v.w);
    if (c4 == 0) { atomicAdd(&deg[u], 1.0f); atomicAdd(&deg[w], 1.0f); }
}

// ---------------------------------------------------------------------------
// Kernel 3: aggr_n[tri_node[t]] += h_e[tri_eid[t]]
// ---------------------------------------------------------------------------
__global__ void aggr_tri_kernel(const float4* __restrict__ he4,
                                const int* __restrict__ tri_node,
                                const int* __restrict__ tri_eid, int T,
                                float* __restrict__ aggr_n) {
    const long idx = (long)blockIdx.x * blockDim.x + threadIdx.x;
    if (idx >= (long)T * (D_ / 4)) return;
    const int t  = (int)(idx >> 5);
    const int c4 = (int)(idx & 31);
    const float4 v = he4[(long)tri_eid[t] * 32 + c4];
    float* p = &aggr_n[(size_t)tri_node[t] * D_ + c4 * 4];
    atomicAdd(p + 0, v.x); atomicAdd(p + 1, v.y);
    atomicAdd(p + 2, v.z); atomicAdd(p + 3, v.w);
}

// ---------------------------------------------------------------------------
// Kernel 4: fused node update (f1 -> masked f2 + h@Wr + br -> p1 -> p2).
// ---------------------------------------------------------------------------
__global__ void __launch_bounds__(128)
node_update_kernel(const float* __restrict__ h,
                   const float* __restrict__ aggr_t, const float* __restrict__ aggr_n,
                   const float* __restrict__ deg, int N,
                   const unsigned int* __restrict__ Wf1s, const float* __restrict__ bf1,
                   const unsigned int* __restrict__ Wf2s, const float* __restrict__ bf2,
                   const unsigned int* __restrict__ Wrs,  const float* __restrict__ br,
                   const unsigned int* __restrict__ Wp1s, const float* __restrict__ bp1,
                   const unsigned int* __restrict__ Wp2s, const float* __restrict__ bp2,
                   float* __restrict__ out) {
    const int LDW = MH_ / 2 + 4;   // 132 dwords (16x256 f16)
    const int LDN = D_ / 2 + 4;    // 68 dwords  (16x128 f16)
    __shared__ __align__(16) unsigned int Xs[16 * (MH_ / 2 + 4)];  // concat; reused for hid2
    __shared__ __align__(16) unsigned int Hs[16 * (MH_ / 2 + 4)];  // hid
    __shared__ __align__(16) unsigned int Ts[16 * (D_ / 2 + 4)];   // h tile
    __shared__ __align__(16) unsigned int Rs[16 * (D_ / 2 + 4)];   // r

    const int node0 = blockIdx.x * 16;
    const int tid   = threadIdx.x;
    const int lane  = tid & 31;
    const int wave  = tid >> 5;

    // Stage X = f16(concat(aggr_t, aggr_n)), Ts = f16(h tile).
    for (int idx = tid; idx < 16 * (MH_ / 2); idx += 128) {
        const int m = idx >> 7, cd = idx & (MH_ / 2 - 1);
        const int node = node0 + m;
        unsigned int pv = 0u;
        if (node < N) {
            const float2* src = (cd < D_ / 2) ? (const float2*)aggr_t
                                              : (const float2*)aggr_n;
            const int c2 = (cd < D_ / 2) ? cd : cd - D_ / 2;
            const float2 a = src[(size_t)node * (D_ / 2) + c2];
            pv = pkh(a.x, a.y);
        }
        Xs[m * LDW + cd] = pv;
    }
    for (int idx = tid; idx < 16 * (D_ / 2); idx += 128) {
        const int m = idx >> 6, cd = idx & (D_ / 2 - 1);
        const int node = node0 + m;
        unsigned int pv = 0u;
        if (node < N) {
            const float2 a = ((const float2*)h)[(size_t)node * (D_ / 2) + cd];
            pv = pkh(a.x, a.y);
        }
        Ts[m * LDN + cd] = pv;
    }
    __syncthreads();

    const int n = lane & 15, half = lane >> 4;

    // f1: hid = relu(X @ Wf1 + bf1)  K=256 -> Hs
    {
        v8f acc[4] = {};
        gemm_loop<4, 8>(acc, Xs, LDW, Wf1s, MH_ / 16, wave * 4, lane);
        _Float16* Hh = (_Float16*)Hs;
#pragma unroll
        for (int t = 0; t < 4; ++t) {
            const int n0 = (wave * 4 + t) * 16;
            const float bias = bf1[n0 + n];
#pragma unroll
            for (int r = 0; r < 8; ++r) {
                const int m = r + 8 * half;
                const float v = acc[t][r] + bias;
                Hh[m * (LDW * 2) + n0 + n] = (_Float16)(v > 0.f ? v : 0.f);
            }
        }
    }
    __syncthreads();

    // f2 (+deg mask), then += h @ Wr + br  -> Rs
    {
        v8f acc[2] = {};
        gemm_loop<2, 8>(acc, Hs, LDW, Wf2s, D_ / 16, wave * 2, lane);
#pragma unroll
        for (int t = 0; t < 2; ++t) {
            const int n0 = (wave * 2 + t) * 16;
            const float bias = bf2[n0 + n];
#pragma unroll
            for (int r = 0; r < 8; ++r) {
                const int m = r + 8 * half;
                const int node = node0 + m;
                const float dg = (node < N) ? deg[node] : 0.f;
                const float zv = acc[t][r] + bias;
                acc[t][r] = (dg == 0.f) ? 0.f : zv;   // mask z BEFORE Wr accum
            }
        }
        gemm_loop<2, 4>(acc, Ts, LDN, Wrs, D_ / 16, wave * 2, lane);
        _Float16* Rh = (_Float16*)Rs;
#pragma unroll
        for (int t = 0; t < 2; ++t) {
            const int n0 = (wave * 2 + t) * 16;
            const float bias = br[n0 + n];
#pragma unroll
            for (int r = 0; r < 8; ++r) {
                const int m = r + 8 * half;
                Rh[m * (LDN * 2) + n0 + n] = (_Float16)(acc[t][r] + bias);
            }
        }
    }
    __syncthreads();

    // p1: hid2 = relu(R @ Wp1 + bp1)  K=128 -> Xs (reuse)
    {
        v8f acc[4] = {};
        gemm_loop<4, 4>(acc, Rs, LDN, Wp1s, MH_ / 16, wave * 4, lane);
        _Float16* Xh = (_Float16*)Xs;
#pragma unroll
        for (int t = 0; t < 4; ++t) {
            const int n0 = (wave * 4 + t) * 16;
            const float bias = bp1[n0 + n];
#pragma unroll
            for (int r = 0; r < 8; ++r) {
                const int m = r + 8 * half;
                const float v = acc[t][r] + bias;
                Xh[m * (LDW * 2) + n0 + n] = (_Float16)(v > 0.f ? v : 0.f);
            }
        }
    }
    __syncthreads();

    // p2: out = hid2 @ Wp2 + bp2  K=256 -> global
    {
        v8f acc[2] = {};
        gemm_loop<2, 8>(acc, Xs, LDW, Wp2s, D_ / 16, wave * 2, lane);
#pragma unroll
        for (int t = 0; t < 2; ++t) {
            const int n0 = (wave * 2 + t) * 16;
            const float bias = bp2[n0 + n];
#pragma unroll
            for (int r = 0; r < 8; ++r) {
                const int m = r + 8 * half;
                const int node = node0 + m;
                if (node < N) out[(size_t)node * D_ + n0 + n] = acc[t][r] + bias;
            }
        }
    }
}

// ---------------------------------------------------------------------------
extern "C" void kernel_launch(void* const* d_in, const int* in_sizes, int n_in,
                              void* d_out, int out_size, void* d_ws, size_t ws_size,
                              hipStream_t stream) {
    const float* h        = (const float*)d_in[0];
    const int*   edges    = (const int*)d_in[1];
    const int*   tri_node = (const int*)d_in[2];
    const int*   tri_eid  = (const int*)d_in[3];
    const float* We1 = (const float*)d_in[4];
    const float* be1 = (const float*)d_in[5];
    const float* We2 = (const float*)d_in[6];
    const float* be2 = (const float*)d_in[7];
    const float* Wf1 = (const float*)d_in[8];
    const float* bf1 = (const float*)d_in[9];
    const float* Wf2 = (const float*)d_in[10];
    const float* bf2 = (const float*)d_in[11];
    const float* Wr  = (const float*)d_in[12];
    const float* br  = (const float*)d_in[13];
    const float* Wp1 = (const float*)d_in[14];
    const float* bp1 = (const float*)d_in[15];
    const float* Wp2 = (const float*)d_in[16];
    const float* bp2 = (const float*)d_in[17];

    const int N = in_sizes[0] / D_;
    const int E = in_sizes[1] / 2;
    const int T = in_sizes[2];

    // Workspace: h_e | aggr_t | aggr_n | deg | 7 swizzled f16 weight buffers
    char*  ws     = (char*)d_ws;
    float* he     = (float*)ws;
    size_t off    = (size_t)E * D_ * sizeof(float);
    float* aggr_t = (float*)(ws + off);  off += (size_t)N * D_ * sizeof(float);
    float* aggr_n = (float*)(ws + off);  off += (size_t)N * D_ * sizeof(float);
    float* deg    = (float*)(ws + off);  off += (size_t)N * sizeof(float);
    off = (off + 31) & ~(size_t)31;

    unsigned int* We1s = (unsigned int*)(ws + off); off += (size_t)D_  * MH_ / 2 * 4;
    unsigned int* We2s = (unsigned int*)(ws + off); off += (size_t)MH_ * D_  / 2 * 4;
    unsigned int* Wf1s = (unsigned int*)(ws + off); off += (size_t)MH_ * MH_ / 2 * 4;
    unsigned int* Wf2s = (unsigned int*)(ws + off); off += (size_t)MH_ * D_  / 2 * 4;
    unsigned int* Wrs  = (unsigned int*)(ws + off); off += (size_t)D_  * D_  / 2 * 4;
    unsigned int* Wp1s = (unsigned int*)(ws + off); off += (size_t)D_  * MH_ / 2 * 4;
    unsigned int* Wp2s = (unsigned int*)(ws + off); off += (size_t)MH_ * D_  / 2 * 4;

    // Zero the accumulation buffers (contiguous), graph-capture safe.
    hipMemsetAsync(aggr_t, 0,
                   (size_t)N * D_ * sizeof(float) * 2 + (size_t)N * sizeof(float),
                   stream);

    // 0) weight swizzle passes (tiny, L2-resident afterwards)
    {
        struct { const float* W; int K, N; unsigned int* O; } jobs[7] = {
            {We1, D_,  MH_, We1s}, {We2, MH_, D_,  We2s},
            {Wf1, MH_, MH_, Wf1s}, {Wf2, MH_, D_,  Wf2s},
            {Wr,  D_,  D_,  Wrs},  {Wp1, D_,  MH_, Wp1s},
            {Wp2, MH_, D_,  Wp2s}};
        for (int j = 0; j < 7; ++j) {
            const int total = (jobs[j].K >> 5) * (jobs[j].N >> 4) * 256;
            swizzle_weight_kernel<<<(total + 255) / 256, 256, 0, stream>>>(
                jobs[j].W, jobs[j].K, jobs[j].N, jobs[j].O);
        }
    }

    // 1) edge MLP (WMMA)
    edge_mlp_kernel<<<(E + 15) / 16, 128, 0, stream>>>(
        h, edges, E, We1s, be1, We2s, be2, he);

    // 2) scatter-add aggregations
    {
        const long work = (long)E * (D_ / 4);
        aggr_edges_kernel<<<(int)((work + 255) / 256), 256, 0, stream>>>(
            (const float4*)he, edges, E, aggr_t, deg);
    }
    {
        const long work = (long)T * (D_ / 4);
        aggr_tri_kernel<<<(int)((work + 255) / 256), 256, 0, stream>>>(
            (const float4*)he, tri_node, tri_eid, T, aggr_n);
    }

    // 3) fused node update (WMMA x5 GEMM chain)
    node_update_kernel<<<(N + 15) / 16, 128, 0, stream>>>(
        h, aggr_t, aggr_n, deg, N,
        Wf1s, bf1, Wf2s, bf2, Wrs, br, Wp1s, bp1, Wp2s, bp2, (float*)d_out);
}
